// BronxLayer_11982958756476
// MI455X (gfx1250) — matrix operand order repaired and checked by reading, not compiled, not run on hardware
//
#include <hip/hip_runtime.h>
#include <math.h>
#include <stdint.h>

typedef __bf16 bf16;
typedef __attribute__((ext_vector_type(16))) __bf16 v16bf;
typedef __attribute__((ext_vector_type(8)))  __bf16 v8bf;
typedef __attribute__((ext_vector_type(4)))  __bf16 v4bf;
typedef __attribute__((ext_vector_type(8)))  float  v8f;
typedef __attribute__((ext_vector_type(4)))  float  v4f;

#define N_NODES 4096
#define HID     256
#define HEADS   4
#define DH      64

// GEMM macro-tile: 128x64 per 256-thread block (8 waves, each 32x32 = 2x2 WMMA tiles)
#define BM 128
#define BN 64
#define BK 32
#define APITCH 40   // halves; 80B rows keep every 16B fragment chunk aligned
#define BPITCH 40

// ---- CDNA5 async global->LDS copy (ASYNCcnt path) -------------------------
__device__ __forceinline__ void async_ld16(unsigned lds_off, const void* gaddr) {
  // per-lane: LDS[lds_off .. +15] = MEM[gaddr .. +15]
  asm volatile("global_load_async_to_lds_b128 %0, %1, off"
               :: "v"(lds_off), "v"(gaddr) : "memory");
}
__device__ __forceinline__ void wait_async0() {
  asm volatile("s_wait_asynccnt 0x0" ::: "memory");
}

// ---------------------------------------------------------------------------
// Generic bf16-WMMA tiled GEMM.  C = A(MxK) * B(KxN), fp32 accumulate.
// A row-major (lda).  AF32=1: f32 source, staged via v4f load + convert.
//                     AF32=0: bf16 source, staged via global_load_async_to_lds.
// BMODE: 0 = generic scalar transpose stage (strides sbk/sbn, f32 or bf16)
//        1 = bf16, n-contiguous (sbn==1): b128 global load + 8 ds_store_b16
//        2 = bf16, K-contiguous rows (sbk==1, row pitch sbn): async copy
// EPI: 0 plain f32 C (per-head stride c_hs), 1 xp interleave (+SHADOW bf16),
//      2 head-gather bf16 (K/Q), 3 head-interleaved O write, 4 elu + f32.
// ---------------------------------------------------------------------------
template<int AF32, int BF32, int BMODE, int EPI, int SHADOW>
__global__ __launch_bounds__(256) void gemm_kernel(
    const void* __restrict__ Ap, long a_hs, int lda,
    const void* __restrict__ Bp, long b_hs, long sbk, long sbn,
    float* __restrict__ C, long c_hs, int ldc,
    bf16* __restrict__ Cb, int epi_p0,
    int M, int Nn, int K)
{
  (void)M; (void)Nn;
  __shared__ bf16 At[BM * APITCH];
  __shared__ bf16 Bt[BN * BPITCH];

  const int tid  = threadIdx.x;
  const int lane = tid & 31;
  const int wave = tid >> 5;
  const int wm   = (wave >> 1) * 32;   // 4 waves along M
  const int wn   = (wave & 1) * 32;    // 2 waves along N
  const int bm   = blockIdx.x * BM;
  const int bn   = blockIdx.y * BN;
  const int hz   = blockIdx.z;         // head index (or 0)

  const float* Af = (const float*)Ap + (size_t)hz * a_hs;
  const bf16*  Ab = (const bf16*) Ap + (size_t)hz * a_hs;
  const float* Bf = (const float*)Bp + (size_t)hz * b_hs;
  const bf16*  Bb = (const bf16*) Bp + (size_t)hz * b_hs;

  const unsigned abase = (unsigned)(uintptr_t)(&At[0]);
  const unsigned bbase = (unsigned)(uintptr_t)(&Bt[0]);

  v8f acc[2][2] = {};

  const int lrow  = lane & 15;
  const int koffA = (lane >> 4) << 3;   // 0 or 8
  const int koffB = (lane >> 4) << 4;   // 0 or 16

  for (int k0 = 0; k0 < K; k0 += BK) {
    __syncthreads();

    // ---- stage A tile [m][k] ----
    if (AF32) {
      // f32 -> bf16, vectorized: 4 elements per chunk
#pragma unroll
      for (int c2 = 0; c2 < (BM * BK / 4) / 256; ++c2) {
        int i = tid + c2 * 256;
        int m = i >> 3, c = i & 7;
        v4f f = *(const v4f*)(Af + (size_t)(bm + m) * lda + (size_t)(k0 + c * 4));
        *(v4bf*)&At[m * APITCH + c * 4] = __builtin_convertvector(f, v4bf);
      }
    } else {
      // async copy, 16B chunks straight into LDS
#pragma unroll
      for (int c2 = 0; c2 < (BM * BK / 8) / 256; ++c2) {
        int i = tid + c2 * 256;
        int m = i >> 2, c = i & 3;
        async_ld16(abase + (unsigned)(m * APITCH + c * 8) * 2u,
                   Ab + (size_t)(bm + m) * lda + (size_t)(k0 + c * 8));
      }
    }

    // ---- stage B tile K-contiguous: Bt[n][k] ----
    if (BMODE == 0) {
      for (int i = tid; i < BK * BN; i += 256) {
        int k = i >> 6, n = i & 63;
        size_t off = (size_t)(k0 + k) * sbk + (size_t)(bn + n) * sbn;
        Bt[n * BPITCH + k] = BF32 ? (bf16)Bf[off] : Bb[off];
      }
    } else if (BMODE == 1) {
      // bf16, contiguous along n: one b128 load, 8 scattered ds stores
      int k = tid >> 3, n8 = (tid & 7) * 8;
      v8bf bv = *(const v8bf*)(Bb + (size_t)(k0 + k) * sbk + (size_t)(bn + n8));
#pragma unroll
      for (int j = 0; j < 8; ++j) Bt[(n8 + j) * BPITCH + k] = bv[j];
    } else {
      // bf16, K-contiguous rows (row pitch sbn): async copy
      int n = tid >> 2, c = tid & 3;
      async_ld16(bbase + (unsigned)(n * BPITCH + c * 8) * 2u,
                 Bb + (size_t)(bn + n) * sbn + (size_t)(k0 + c * 8));
    }

    if (!AF32 || BMODE == 2) wait_async0();
    __syncthreads();

    v16bf afrag[2], bfrag[2];
#pragma unroll
    for (int t = 0; t < 2; ++t) {
      // A 16x32 bf16 fragment: lanes 0-15 -> K 0-7 / 16-23, lanes 16-31 -> K 8-15 / 24-31
      const bf16* pa = &At[(wm + t * 16 + lrow) * APITCH];
      v8bf a0 = *(const v8bf*)(pa + koffA);
      v8bf a1 = *(const v8bf*)(pa + 16 + koffA);
      afrag[t] = __builtin_shufflevector(a0, a1, 0,1,2,3,4,5,6,7,8,9,10,11,12,13,14,15);
      // B 32x16 bf16 fragment: N = lane%16, lanes 0-15 -> K 0-7 / 16-23, lanes 16-31 -> K 8-15 / 24-31
      const bf16* pb = &Bt[(wn + t * 16 + lrow) * BPITCH];
      v8bf b0 = *(const v8bf*)(pb + koffA);
      v8bf b1 = *(const v8bf*)(pb + 16 + koffA);
      bfrag[t] = __builtin_shufflevector(b0, b1, 0,1,2,3,4,5,6,7,8,9,10,11,12,13,14,15);
    }
#pragma unroll
    for (int i = 0; i < 2; ++i)
#pragma unroll
      for (int j = 0; j < 2; ++j)
        acc[i][j] = __builtin_amdgcn_wmma_f32_16x16x32_bf16(
            false, afrag[i], false, bfrag[j], (short)0, acc[i][j], false, false);
  }

  // epilogue: C/D layout -> VGPR r, lanes 0-15: M=r, lanes 16-31: M=r+8; N=lane%16
#pragma unroll
  for (int i = 0; i < 2; ++i) {
#pragma unroll
    for (int j = 0; j < 2; ++j) {
#pragma unroll
      for (int r = 0; r < 8; ++r) {
        int row = bm + wm + i * 16 + r + ((lane >> 4) << 3);
        int col = bn + wn + j * 16 + (lane & 15);
        float v = acc[i][j][r];
        if (EPI == 0) {
          C[(size_t)hz * c_hs + (size_t)row * ldc + col] = v;
        } else if (EPI == 1) {
          C[((size_t)row * N_NODES + col) * HEADS + epi_p0] = v;   // xp[x,z,b]
          if (SHADOW) Cb[(size_t)row * N_NODES + col] = (bf16)v;   // bf16 shadow
        } else if (EPI == 2) {
          // column c of hn@W maps to head b=c&3, d=c>>2 -> head-major [b][n][d]
          Cb[(size_t)(col & 3) * (N_NODES * DH) + (size_t)row * DH + (col >> 2)] = (bf16)v;
        } else if (EPI == 3) {
          C[(size_t)row * HID + col * HEADS + hz] = v;             // O[n, d*4+b]
        } else {
          float e = v > 0.0f ? v : (expf(v) - 1.0f);               // elu
          C[(size_t)row * ldc + col] = e;
        }
      }
    }
  }
}

// ---------------------------------------------------------------------------
// Row L1-normalize x -> xn (bf16) and init xp[...,0]=I, xp[...,1]=xn (f32)
// ---------------------------------------------------------------------------
__global__ __launch_bounds__(256) void rownorm_kernel(
    const float* __restrict__ x, bf16* __restrict__ xnb, float* __restrict__ xp)
{
  const int row = blockIdx.x;
  const int tid = threadIdx.x;
  const float* xr = x + (size_t)row * N_NODES;
  __shared__ float red[256];
  float s = 0.f;
  for (int z = tid; z < N_NODES; z += 256) s += fabsf(xr[z]);
  red[tid] = s; __syncthreads();
  for (int off = 128; off; off >>= 1) { if (tid < off) red[tid] += red[tid + off]; __syncthreads(); }
  const float inv = 1.0f / fmaxf(red[0], 1e-12f);
  for (int z = tid; z < N_NODES; z += 256) {
    float v = xr[z] * inv;
    xnb[(size_t)row * N_NODES + z] = (bf16)v;
    size_t o = ((size_t)row * N_NODES + z) * HEADS;
    xp[o + 0] = (row == z) ? 1.0f : 0.0f;
    xp[o + 1] = v;
  }
}

// ---------------------------------------------------------------------------
// LayerNorm over HID=256 (one block per row, one element per thread)
// ---------------------------------------------------------------------------
__global__ __launch_bounds__(256) void layernorm_kernel(
    const float* __restrict__ h, const float* __restrict__ gamma,
    const float* __restrict__ beta, float* __restrict__ hn)
{
  const int row = blockIdx.x;
  const int tid = threadIdx.x;
  float v = h[(size_t)row * HID + tid];
  __shared__ float red[256];
  red[tid] = v; __syncthreads();
  for (int off = 128; off; off >>= 1) { if (tid < off) red[tid] += red[tid + off]; __syncthreads(); }
  const float mu = red[0] * (1.0f / HID); __syncthreads();
  const float d = v - mu;
  red[tid] = d * d; __syncthreads();
  for (int off = 128; off; off >>= 1) { if (tid < off) red[tid] += red[tid + off]; __syncthreads(); }
  const float var = red[0] * (1.0f / HID);
  hn[(size_t)row * HID + tid] = d * rsqrtf(var + 1e-5f) * gamma[tid] + beta[tid];
}

// ---------------------------------------------------------------------------
// a = softmax_z(S); a = softmax_z(a * xp[:,:,b]); result written as bf16 into
// the FRONT HALF of this row's own 16KB region (no cross-block aliasing), so
// the aggregation GEMM can run on the bf16/async/WMMA path (lda = 8192).
// ---------------------------------------------------------------------------
__global__ __launch_bounds__(256) void softmax2_kernel(
    float* __restrict__ S, const float* __restrict__ xp)
{
  const int x = blockIdx.x;
  const int b = blockIdx.y;
  const int tid = threadIdx.x;
  float* row = S + ((size_t)b * N_NODES + x) * N_NODES;
  __shared__ float red[256];

  float v[16];
  float m = -1e30f;
#pragma unroll
  for (int i = 0; i < 16; ++i) { v[i] = row[tid + i * 256]; m = fmaxf(m, v[i]); }
  red[tid] = m; __syncthreads();
  for (int off = 128; off; off >>= 1) { if (tid < off) red[tid] = fmaxf(red[tid], red[tid + off]); __syncthreads(); }
  m = red[0]; __syncthreads();
  float s = 0.f;
#pragma unroll
  for (int i = 0; i < 16; ++i) { v[i] = expf(v[i] - m); s += v[i]; }
  red[tid] = s; __syncthreads();
  for (int off = 128; off; off >>= 1) { if (tid < off) red[tid] += red[tid + off]; __syncthreads(); }
  float invZ = 1.0f / red[0]; __syncthreads();

  m = -1e30f;
#pragma unroll
  for (int i = 0; i < 16; ++i) {
    int z = tid + i * 256;
    v[i] = v[i] * invZ * xp[((size_t)x * N_NODES + z) * HEADS + b];
    m = fmaxf(m, v[i]);
  }
  red[tid] = m; __syncthreads();
  for (int off = 128; off; off >>= 1) { if (tid < off) red[tid] = fmaxf(red[tid], red[tid + off]); __syncthreads(); }
  m = red[0]; __syncthreads();
  s = 0.f;
#pragma unroll
  for (int i = 0; i < 16; ++i) { v[i] = expf(v[i] - m); s += v[i]; }
  red[tid] = s; __syncthreads();
  for (int off = 128; off; off >>= 1) { if (tid < off) red[tid] += red[tid + off]; __syncthreads(); }
  invZ = 1.0f / red[0];
  __syncthreads();                  // all reads of this row are done
  bf16* brow = (bf16*)row;          // in-place bf16 result, elements 0..4095
#pragma unroll
  for (int i = 0; i < 16; ++i) brow[tid + i * 256] = (bf16)(v[i] * invZ);
}

// ---------------------------------------------------------------------------
extern "C" void kernel_launch(void* const* d_in, const int* in_sizes, int n_in,
                              void* d_out, int out_size, void* d_ws, size_t ws_size,
                              hipStream_t stream)
{
  (void)in_sizes; (void)n_in; (void)out_size; (void)ws_size;
  const float* h     = (const float*)d_in[0];
  const float* x     = (const float*)d_in[1];
  const float* w_k   = (const float*)d_in[2];
  const float* w_q   = (const float*)d_in[3];
  const float* w_fc  = (const float*)d_in[4];
  const float* gamma = (const float*)d_in[5];
  const float* beta  = (const float*)d_in[6];

  float* out = (float*)d_out;                          // [4096, 1024]
  float* xp  = out + (size_t)N_NODES * HID * HEADS;    // [4096, 4096, 4]

  char* w = (char*)d_ws;
  bf16*  xnb = (bf16*)w;  w += (size_t)N_NODES * N_NODES * 2;        // 32 MB
  bf16*  p2b = (bf16*)w;  w += (size_t)N_NODES * N_NODES * 2;        // 32 MB
  float* hn  = (float*)w; w += (size_t)N_NODES * HID * 4;            //  4 MB
  bf16*  kh  = (bf16*)w;  w += (size_t)HEADS * N_NODES * DH * 2;     //  2 MB
  bf16*  qh  = (bf16*)w;  w += (size_t)HEADS * N_NODES * DH * 2;     //  2 MB
  float* Sb  = (float*)w; w += (size_t)HEADS * N_NODES * N_NODES * 4;// 256 MB
  float* O   = (float*)w; w += (size_t)N_NODES * HID * 4;            //  4 MB

  rownorm_kernel<<<N_NODES, 256, 0, stream>>>(x, xnb, xp);
  layernorm_kernel<<<N_NODES, 256, 0, stream>>>(h, gamma, beta, hn);

  dim3 gP(N_NODES / BM, N_NODES / BN);
  // xp[...,2] = xn@xn (+ bf16 shadow for the next power)
  gemm_kernel<0,0,1,1,1><<<gP, 256, 0, stream>>>(xnb, 0, N_NODES, xnb, 0, N_NODES, 1,
                                                 xp, 0, 0, p2b, 2, N_NODES, N_NODES, N_NODES);
  // xp[...,3] = (xn^2)@xn
  gemm_kernel<0,0,1,1,0><<<gP, 256, 0, stream>>>(p2b, 0, N_NODES, xnb, 0, N_NODES, 1,
                                                 xp, 0, 0, nullptr, 3, N_NODES, N_NODES, N_NODES);
  // head-gathered K/Q (bf16, head-major [b][n][d])
  dim3 gKQ(N_NODES / BM, HID / BN);
  gemm_kernel<1,1,0,2,0><<<gKQ, 256, 0, stream>>>(hn, 0, HID, w_k, 0, HID, 1,
                                                  nullptr, 0, 0, kh, 0, N_NODES, HID, HID);
  gemm_kernel<1,1,0,2,0><<<gKQ, 256, 0, stream>>>(hn, 0, HID, w_q, 0, HID, 1,
                                                  nullptr, 0, 0, qh, 0, N_NODES, HID, HID);
  // S_b = kh_b @ qh_b^T : B is K-contiguous (row pitch DH) -> async B path
  dim3 gS(N_NODES / BM, N_NODES / BN, HEADS);
  gemm_kernel<0,0,2,0,0><<<gS, 256, 0, stream>>>(kh, (long)N_NODES * DH, DH,
                                                 qh, (long)N_NODES * DH, 1, DH,
                                                 Sb, (long)N_NODES * N_NODES, N_NODES,
                                                 nullptr, 0, N_NODES, N_NODES, DH);
  // double softmax with xp reweighting; bf16 result in place (lda 8192)
  softmax2_kernel<<<dim3(N_NODES, HEADS), 256, 0, stream>>>(Sb, xp);
  // O_b = a2_b @ hh_b   (A bf16 in-place, head stride 4096*8192 bf16 elements)
  dim3 gO(N_NODES / BM, DH / BN, HEADS);
  gemm_kernel<0,1,0,3,0><<<gO, 256, 0, stream>>>(Sb, (long)N_NODES * 2 * N_NODES, 2 * N_NODES,
                                                 hn, 1, HID, HEADS,
                                                 O, 0, 0, nullptr, 0, N_NODES, DH, N_NODES);
  // out = elu(O @ w_fc)
  dim3 gF(N_NODES / BM, (HID * HEADS) / BN);
  gemm_kernel<1,1,0,4,0><<<gF, 256, 0, stream>>>(O, 0, HID, w_fc, 0, HID * HEADS, 1,
                                                 out, 0, HID * HEADS, nullptr, 0,
                                                 N_NODES, HID * HEADS, HID);
}
